// LinearWindowToFilters_48473000903619
// MI455X (gfx1250) — matrix operand order repaired and checked by reading, not compiled, not run on hardware
//
#include <hip/hip_runtime.h>
#include <math.h>

// ---------------------------------------------------------------------------
// MLP funnel 1024->130->110->90->70->50->64 over B=131072 rows on MI455X.
// fp16 WMMA GEMM (v_wmma_f32_16x16x32_f16), fp32 accumulate.
// All padding / BN-affine / ReLU / f32->f16 conversion happens in tiny pack
// kernels that write WMMA-native fragment layouts, so the GEMM inner loop is
// pure: unconditional b128 loads + v_wmma, no exec-mask branching.
// ---------------------------------------------------------------------------

typedef __attribute__((ext_vector_type(16))) _Float16 v16h;
typedef __attribute__((ext_vector_type(8)))  float    v8f;

#define BATCH   131072
#define MTILES  (BATCH / 16)          // 8192
#define WAVES_PER_BLOCK 8

// ---------------- zero-fill helper -----------------------------------------
__global__ void zero_kernel(float* p, int n) {
  int i = blockIdx.x * blockDim.x + threadIdx.x;
  if (i < n) p[i] = 0.0f;
}

// ---------------- pad bias to Npad (guard-free epilogue) --------------------
__global__ void pack_bias(const float* __restrict__ bias, float* __restrict__ bpad,
                          int N_real, int Npad) {
  int j = threadIdx.x;
  if (j < Npad) bpad[j] = (j < N_real) ? bias[j] : 0.0f;
}

// ---------------- pack weights into WMMA B-fragment layout ------------------
// Wpack[(kt*NT + t)*32 + lane][16 f16]; B layout: N = lane%16, element j within
// the fragment maps to kk = kt*32 + (j/8)*16 + half*8 + j%8.
__global__ void pack_weights(const float* __restrict__ Wt, _Float16* __restrict__ Wpack,
                             int K_real, int N_real, int NT, int kticks) {
  int tid = blockIdx.x * blockDim.x + threadIdx.x;
  int total = kticks * NT * 32;
  if (tid >= total) return;
  int lane = tid & 31, frag = tid >> 5;
  int t = frag % NT, kt = frag / NT;
  int nlane = lane & 15, half = lane >> 4;
  int ncol = t * 16 + nlane;
  int ncolc = (ncol < N_real) ? ncol : 0;          // clamped (branch-free)
  v16h out;
#pragma unroll
  for (int j = 0; j < 16; ++j) {
    int kk  = kt * 32 + ((j >> 3) << 4) + (half << 3) + (j & 7);
    int kkc = (kk < K_real) ? kk : 0;
    float w = Wt[(size_t)ncolc * K_real + kkc];
    out[j] = (_Float16)((ncol < N_real && kk < K_real) ? w : 0.0f);
  }
  *(v16h*)(Wpack + (size_t)tid * 16) = out;
}

// ---------------- BN affine + ReLU + cvt + pack into A-fragment layout ------
// Reads tile-major Y of the previous layer:
//   Y[((mtile*NT_PREV + n/16)*32 + (m%16)/8*16 + n%16)*8 + m%8]
// Writes Xpack[(mtile*kticks + kt)*32 + lane][16 f16] in the ISA A layout:
//   element j: v=j/2, kk = kt*32 + (v/4)*16 + half*8 + (v%4)*2 + j%2.
template <int K_REAL, int NT_PREV>
__global__ __launch_bounds__(256) void act_pack(
    const float* __restrict__ Ytiled,
    const float* __restrict__ scale, const float* __restrict__ shift,
    _Float16* __restrict__ Xpack, int kticks)
{
  int tid  = blockIdx.x * blockDim.x + threadIdx.x;
  int lane = tid & 31, frag = tid >> 5;
  int kt = frag % kticks, mtile = frag / kticks;
  int arow = lane & 15, half = lane >> 4;
  int hp = arow >> 3, r = arow & 7;
  v16h out;
#pragma unroll
  for (int j = 0; j < 16; ++j) {
    int v  = j >> 1;
    int kk = kt * 32 + ((v >> 2) << 4) + (half << 3) + ((v & 3) << 1) + (j & 1);
    int kkc = (kk < K_REAL) ? kk : 0;              // clamp -> no OOB, no branch
    size_t idx = ((size_t)(mtile * NT_PREV + (kkc >> 4)) * 32 + hp * 16 + (kkc & 15)) * 8 + r;
    float x = Ytiled[idx];
    float a = fmaxf(0.0f, fmaf(scale[kkc], x, shift[kkc]));
    out[j] = (_Float16)((kk < K_REAL) ? a : 0.0f);
  }
  *(v16h*)(Xpack + (size_t)tid * 16) = out;
}

// ---------------- WMMA GEMM + column stats ----------------------------------
// One wave = one 16-row M tile x NT 16-col N tiles. Inner loop: unconditional
// 32B fragment loads + v_wmma. Output stored tile-major (v8f per lane).
template <int NT, bool APACK>
__global__ __launch_bounds__(256) void gemm_bn(
    const float* __restrict__ Xf32, int ldx,        // APACK=false path
    const _Float16* __restrict__ Xpack,             // APACK=true path
    const _Float16* __restrict__ Wpack,             // [kticks][NT][32][16]
    const float* __restrict__ bpad,                 // [NT*16], zero-padded
    float* __restrict__ Ytiled,                     // [mtiles][NT][32][8]
    float* __restrict__ csum, float* __restrict__ csq, int do_stats,
    int kticks)
{
  const int lane  = threadIdx.x & 31;
  const int wave  = threadIdx.x >> 5;
  const int mtile = blockIdx.x * WAVES_PER_BLOCK + wave;
  const int half  = lane >> 4;
  const int nlane = lane & 15;

  v8f acc[NT];
#pragma unroll
  for (int t = 0; t < NT; ++t)
#pragma unroll
    for (int r = 0; r < 8; ++r) acc[t][r] = 0.0f;

  const _Float16* aptr = Xpack + ((size_t)mtile * kticks * 32 + lane) * 16;
  const float*    xrow = Xf32 + (size_t)(mtile * 16 + nlane) * ldx;
  const _Float16* bptr = Wpack + (size_t)lane * 16;

  for (int kt = 0; kt < kticks; ++kt) {
    v16h a;
    if constexpr (APACK) {
      a = *(const v16h*)aptr;
      aptr += 32 * 16;
    } else {
      // K exact multiple of 32: unconditional paired fp32 loads + cvt.
      const float* xr = xrow + kt * 32;
#pragma unroll
      for (int v = 0; v < 8; ++v) {
        const int kkoff = ((v >> 2) << 4) + (half << 3) + ((v & 3) << 1);
        float2 xv = *(const float2*)(xr + kkoff);
        a[2 * v]     = (_Float16)xv.x;
        a[2 * v + 1] = (_Float16)xv.y;
      }
    }
#pragma unroll
    for (int t = 0; t < NT; ++t) {
      v16h b = *(const v16h*)(bptr + (size_t)t * 512);
      acc[t] = __builtin_amdgcn_wmma_f32_16x16x32_f16(
          false, a, false, b, (short)0, acc[t], false, false);
    }
    bptr += (size_t)NT * 512;
  }

  // ---- epilogue: bias, tile-major store, column stats ----------------------
#pragma unroll
  for (int t = 0; t < NT; ++t) {
    const int ncol = t * 16 + nlane;
    const float bv = bpad[ncol];
    v8f yv;
    float s = 0.0f, s2 = 0.0f;
#pragma unroll
    for (int r = 0; r < 8; ++r) {
      float y = acc[t][r] + bv;
      yv[r] = y;
      s += y; s2 += y * y;
    }
    *(v8f*)(Ytiled + ((size_t)(mtile * NT + t) * 32 + lane) * 8) = yv;
    if (do_stats) {
      s  += __shfl_xor(s,  16, 32);   // fold the two half-lanes (same column)
      s2 += __shfl_xor(s2, 16, 32);
      if (half == 0) {                // padded columns contribute exactly 0
        atomicAdd(&csum[ncol], s);
        atomicAdd(&csq[ncol],  s2);
      }
    }
  }
}

// ---------------- fold column sums into BN affine scale/shift ---------------
__global__ void bn_stats(const float* __restrict__ csum, const float* __restrict__ csq,
                         const float* __restrict__ g, const float* __restrict__ be,
                         float* __restrict__ scale, float* __restrict__ shift,
                         int N_real, int Npad, float invB)
{
  int j = blockIdx.x * blockDim.x + threadIdx.x;
  if (j >= Npad) return;
  if (j < N_real) {
    float mu  = csum[j] * invB;
    float var = csq[j]  * invB - mu * mu;       // biased var, matches jnp.var
    float inv = rsqrtf(var + 1e-5f);
    float sc  = g[j] * inv;
    scale[j] = sc;
    shift[j] = be[j] - mu * sc;
  } else {
    scale[j] = 0.0f;
    shift[j] = 0.0f;
  }
}

// ---------------- jax threefry2x32 (key = (0, 42)) --------------------------
__device__ __forceinline__ unsigned rotl32(unsigned x, int r) {
  return (x << r) | (x >> (32 - r));
}
__device__ __forceinline__ void threefry2x32(unsigned k0, unsigned k1,
                                             unsigned x0, unsigned x1,
                                             unsigned& o0, unsigned& o1) {
  const unsigned ks2 = k0 ^ k1 ^ 0x1BD11BDAu;
  const int RA[4] = {13, 15, 26, 6};
  const int RB[4] = {17, 29, 16, 24};
  x0 += k0; x1 += k1;
#pragma unroll
  for (int i = 0; i < 4; ++i) { x0 += x1; x1 = rotl32(x1, RA[i]); x1 ^= x0; }
  x0 += k1; x1 += ks2 + 1u;
#pragma unroll
  for (int i = 0; i < 4; ++i) { x0 += x1; x1 = rotl32(x1, RB[i]); x1 ^= x0; }
  x0 += ks2; x1 += k0 + 2u;
#pragma unroll
  for (int i = 0; i < 4; ++i) { x0 += x1; x1 = rotl32(x1, RA[i]); x1 ^= x0; }
  x0 += k0; x1 += k1 + 3u;
#pragma unroll
  for (int i = 0; i < 4; ++i) { x0 += x1; x1 = rotl32(x1, RB[i]); x1 ^= x0; }
  x0 += k1; x1 += ks2 + 4u;
#pragma unroll
  for (int i = 0; i < 4; ++i) { x0 += x1; x1 = rotl32(x1, RA[i]); x1 ^= x0; }
  x0 += ks2; x1 += k0 + 5u;
  o0 = x0; o1 = x1;
}

// ---------------- sigmoid + Bernoulli sample + log-prob row reduce ----------
// Z is tile-major with NT=4 (64 logits per row).
__global__ __launch_bounds__(256) void sample_kernel(
    const float* __restrict__ Ztiled,
    float* __restrict__ mask_out,     // [B x 64]
    float* __restrict__ logp_out,     // [B]
    int Brows)
{
  const int lane = threadIdx.x & 31;
  const int wave = threadIdx.x >> 5;
  const int row  = blockIdx.x * WAVES_PER_BLOCK + wave;
  if (row >= Brows) return;

  const int mtile = row >> 4, mrow = row & 15;
  const int hp = mrow >> 3, r = mrow & 7;
  const unsigned halfn = (unsigned)Brows * 32u;  // (B*64)/2
  float lg = 0.0f;
#pragma unroll
  for (int c2 = 0; c2 < 2; ++c2) {
    const int col = lane + 32 * c2;
    const size_t idx = ((size_t)(mtile * 4 + (col >> 4)) * 32 + hp * 16 + (col & 15)) * 8 + r;
    const float z = Ztiled[idx];
    const float p = 1.0f / (1.0f + expf(-z));

    const unsigned e = (unsigned)row * 64u + (unsigned)col;
    unsigned i, pick_lo, o0, o1;
    if (e < halfn) { i = e;         pick_lo = 1u; }
    else           { i = e - halfn; pick_lo = 0u; }
    threefry2x32(0u, 42u, i, i + halfn, o0, o1);
    const unsigned bits = pick_lo ? o0 : o1;
    const float u = __uint_as_float((bits >> 9) | 0x3f800000u) - 1.0f;

    const float chosen = (u < 1.0f - p) ? 1.0f : 0.0f;
    mask_out[(size_t)row * 64 + col] = 1.0f - chosen;
    lg += logf(fabsf(chosen - p));
  }
  lg += __shfl_xor(lg, 1,  32);
  lg += __shfl_xor(lg, 2,  32);
  lg += __shfl_xor(lg, 4,  32);
  lg += __shfl_xor(lg, 8,  32);
  lg += __shfl_xor(lg, 16, 32);
  if (lane == 0) logp_out[row] = lg;
}

// ---------------------------------------------------------------------------
extern "C" void kernel_launch(void* const* d_in, const int* in_sizes, int n_in,
                              void* d_out, int out_size, void* d_ws, size_t ws_size,
                              hipStream_t stream) {
  (void)in_sizes; (void)n_in; (void)out_size; (void)ws_size;
  const float* events = (const float*)d_in[0];
  const int B = BATCH;
  const float invB = 1.0f / (float)B;

  // Layer geometry: {K_real, kticks, N_real, NT}
  // L0: 1024/32/130/9  L1: 130/5/110/7  L2: 110/4/90/6
  // L3: 90/3/70/5      L4: 70/3/50/4    L5: 50/2/64/4
  const int   Kr[6] = {1024, 130, 110, 90, 70, 50};
  const int   kt_[6] = {32, 5, 4, 3, 3, 2};
  const int   Nr[6] = {130, 110, 90, 70, 50, 64};
  const int   NTs[6] = {9, 7, 6, 5, 4, 4};

  // ---- workspace layout (floats) ----
  float* ws = (float*)d_ws;
  size_t o = 0;
  float* Y0 = ws + o; o += (size_t)B * 144;
  float* Y1 = ws + o; o += (size_t)B * 112;
  float* Y2 = ws + o; o += (size_t)B * 96;
  float* Y3 = ws + o; o += (size_t)B * 80;
  float* Y4 = ws + o; o += (size_t)B * 64;
  float* Zb = ws + o; o += (size_t)B * 64;
  // shared A-fragment buffer (max kticks among layers 1..5 is 5)
  _Float16* Xpack = (_Float16*)(ws + o); o += (size_t)MTILES * 5 * 512 / 2;
  // packed weights per layer
  _Float16* Wp[6];
  for (int l = 0; l < 6; ++l) {
    Wp[l] = (_Float16*)(ws + o);
    o += ((size_t)kt_[l] * NTs[l] * 512 + 1) / 2;
  }
  float* bpad_all  = ws + o; o += 6 * 160;
  float* csum_all  = ws + o; o += 5 * 160;
  float* csq_all   = ws + o; o += 5 * 160;
  float* scale_all = ws + o; o += 5 * 160;
  float* shift_all = ws + o; o += 5 * 160;

  // ---- zero atomic accumulators (deterministic, capture-safe) ----
  zero_kernel<<<dim3((5 * 160 * 2 + 255) / 256), dim3(256), 0, stream>>>(
      csum_all, 5 * 160 * 2);

  // ---- pack weights + pad biases ----
  for (int l = 0; l < 6; ++l) {
    const float* W = (const float*)d_in[l < 5 ? (1 + 4 * l) : 21];
    const float* b = (const float*)d_in[l < 5 ? (2 + 4 * l) : 22];
    int total = kt_[l] * NTs[l] * 32;
    pack_weights<<<dim3((total + 255) / 256), dim3(256), 0, stream>>>(
        W, Wp[l], Kr[l], Nr[l], NTs[l], kt_[l]);
    pack_bias<<<dim3(1), dim3(160), 0, stream>>>(b, bpad_all + l * 160, Nr[l], 160);
  }

  const dim3 ggrid(B / (16 * WAVES_PER_BLOCK));  // 1024 blocks
  const dim3 gblk(256);

  // ---- layer 0: events (fp32 direct) -> Y0 ----
  gemm_bn<9, false><<<ggrid, gblk, 0, stream>>>(
      events, 1024, (const _Float16*)nullptr, Wp[0], bpad_all + 0 * 160,
      Y0, csum_all + 0 * 160, csq_all + 0 * 160, 1, 32);
  bn_stats<<<dim3(1), dim3(160), 0, stream>>>(
      csum_all + 0 * 160, csq_all + 0 * 160,
      (const float*)d_in[3], (const float*)d_in[4],
      scale_all + 0 * 160, shift_all + 0 * 160, 130, 160, invB);

  // ---- layer 1 ----
  act_pack<130, 9><<<dim3(1024 * 5), gblk, 0, stream>>>(
      Y0, scale_all + 0 * 160, shift_all + 0 * 160, Xpack, 5);
  gemm_bn<7, true><<<ggrid, gblk, 0, stream>>>(
      nullptr, 0, Xpack, Wp[1], bpad_all + 1 * 160,
      Y1, csum_all + 1 * 160, csq_all + 1 * 160, 1, 5);
  bn_stats<<<dim3(1), dim3(160), 0, stream>>>(
      csum_all + 1 * 160, csq_all + 1 * 160,
      (const float*)d_in[7], (const float*)d_in[8],
      scale_all + 1 * 160, shift_all + 1 * 160, 110, 160, invB);

  // ---- layer 2 ----
  act_pack<110, 7><<<dim3(1024 * 4), gblk, 0, stream>>>(
      Y1, scale_all + 1 * 160, shift_all + 1 * 160, Xpack, 4);
  gemm_bn<6, true><<<ggrid, gblk, 0, stream>>>(
      nullptr, 0, Xpack, Wp[2], bpad_all + 2 * 160,
      Y2, csum_all + 2 * 160, csq_all + 2 * 160, 1, 4);
  bn_stats<<<dim3(1), dim3(160), 0, stream>>>(
      csum_all + 2 * 160, csq_all + 2 * 160,
      (const float*)d_in[11], (const float*)d_in[12],
      scale_all + 2 * 160, shift_all + 2 * 160, 90, 160, invB);

  // ---- layer 3 ----
  act_pack<90, 6><<<dim3(1024 * 3), gblk, 0, stream>>>(
      Y2, scale_all + 2 * 160, shift_all + 2 * 160, Xpack, 3);
  gemm_bn<5, true><<<ggrid, gblk, 0, stream>>>(
      nullptr, 0, Xpack, Wp[3], bpad_all + 3 * 160,
      Y3, csum_all + 3 * 160, csq_all + 3 * 160, 1, 3);
  bn_stats<<<dim3(1), dim3(160), 0, stream>>>(
      csum_all + 3 * 160, csq_all + 3 * 160,
      (const float*)d_in[15], (const float*)d_in[16],
      scale_all + 3 * 160, shift_all + 3 * 160, 70, 160, invB);

  // ---- layer 4 ----
  act_pack<70, 5><<<dim3(1024 * 3), gblk, 0, stream>>>(
      Y3, scale_all + 3 * 160, shift_all + 3 * 160, Xpack, 3);
  gemm_bn<4, true><<<ggrid, gblk, 0, stream>>>(
      nullptr, 0, Xpack, Wp[4], bpad_all + 4 * 160,
      Y4, csum_all + 4 * 160, csq_all + 4 * 160, 1, 3);
  bn_stats<<<dim3(1), dim3(160), 0, stream>>>(
      csum_all + 4 * 160, csq_all + 4 * 160,
      (const float*)d_in[19], (const float*)d_in[20],
      scale_all + 4 * 160, shift_all + 4 * 160, 50, 160, invB);

  // ---- layer 5: logits (no stats) ----
  act_pack<50, 4><<<dim3(1024 * 2), gblk, 0, stream>>>(
      Y4, scale_all + 4 * 160, shift_all + 4 * 160, Xpack, 2);
  gemm_bn<4, true><<<ggrid, gblk, 0, stream>>>(
      nullptr, 0, Xpack, Wp[5], bpad_all + 5 * 160,
      Zb, csum_all, csq_all, 0, 2);

  // ---- sigmoid + threefry sample + log-prob ----
  float* mask_out = (float*)d_out;                   // [B x 64]
  float* logp_out = (float*)d_out + (size_t)B * 64;  // [B]
  sample_kernel<<<dim3(B / WAVES_PER_BLOCK), dim3(256), 0, stream>>>(
      Zb, mask_out, logp_out, B);
}
